// EnhancedTherapeuticGNN_20229295964569
// MI455X (gfx1250) — compile-verified, matrix operand order checked by reading, not run on hardware
//
#include <hip/hip_runtime.h>
#include <hip/hip_bf16.h>

typedef __attribute__((ext_vector_type(16))) __bf16 v16bf;
typedef __attribute__((ext_vector_type(8)))  float  v8f;

#define HID 64
#define NEG_SLOPE 0.2f

// ---------- helpers ----------
__device__ __forceinline__ unsigned fenc(float f) {
    unsigned b = __float_as_uint(f);
    return (b & 0x80000000u) ? ~b : (b | 0x80000000u);
}
__device__ __forceinline__ float fdec(unsigned u) {
    unsigned b = (u & 0x80000000u) ? (u & 0x7FFFFFFFu) : ~u;
    return __uint_as_float(b);
}
__device__ __forceinline__ void edge_sd(const int* __restrict__ ei, int E, int e,
                                        int& s, int& d) {
    if (e < E) { s = ei[e]; d = ei[E + e]; }
    else       { s = e - E; d = e - E; }          // appended self-loop
}
__device__ __forceinline__ float edge_e(const float* __restrict__ As,
                                        const float* __restrict__ Ad,
                                        int s, int d) {
    float v = As[s] + Ad[d];
    return v > 0.0f ? v : NEG_SLOPE * v;          // leaky relu
}

// ---------- pack W[K x 64] (f32, row-major) into bf16 B-fragments ----------
// layout: frag[ntile][kstep][lane][16]; lane holds col n=t*16+(lane&15),
// k = s*32 + ((lane>>4)<<3) + {0..7} and +16..+23  (16-bit B 32x16 layout)
__global__ void pack_w_bf16(const float* __restrict__ W, __bf16* __restrict__ out, int K) {
    const int KS = K >> 5;
    int id = blockIdx.x * blockDim.x + threadIdx.x;
    if (id >= 4 * KS * 32) return;
    int lane = id & 31;
    int s    = (id >> 5) % KS;
    int t    = id / (32 * KS);
    int n  = t * 16 + (lane & 15);
    int kb = s * 32 + ((lane >> 4) << 3);
    v16bf frag;
#pragma unroll
    for (int i = 0; i < 8; ++i) {
        frag[i]     = (__bf16)W[(size_t)(kb + i)      * HID + n];
        frag[8 + i] = (__bf16)W[(size_t)(kb + 16 + i) * HID + n];
    }
    ((v16bf*)out)[(t * KS + s) * 32 + lane] = frag;
}

// ---------- H[N x 64] = X[N x K] @ W (packed bf16)  via v_wmma_f32_16x16x32_bf16 ----
__global__ void gat_gemm_bf16(const float* __restrict__ X,
                              const __bf16* __restrict__ Bpack,
                              float* __restrict__ H, int N, int K) {
    const int KS   = K >> 5;
    const int lane = threadIdx.x & 31;
    const int mtile = (blockIdx.x * blockDim.x + threadIdx.x) >> 5; // one wave = 16x64 tile
    if (mtile * 16 >= N) return;                                    // whole-wave exit (EXEC stays full)

    const int mrow  = mtile * 16 + (lane & 15);
    const int khalf = (lane >> 4) << 3;
    const v16bf* bp = (const v16bf*)Bpack;

    v8f acc0 = {}, acc1 = {}, acc2 = {}, acc3 = {};
    for (int s = 0; s < KS; ++s) {
        const float* xp = X + (size_t)mrow * K + s * 32 + khalf;
        v16bf a;
#pragma unroll
        for (int i = 0; i < 8; ++i) {
            a[i]     = (__bf16)xp[i];       // 16-bit A 16x32 lane layout
            a[8 + i] = (__bf16)xp[16 + i];
        }
        v16bf b0 = bp[(0 * KS + s) * 32 + lane];
        v16bf b1 = bp[(1 * KS + s) * 32 + lane];
        v16bf b2 = bp[(2 * KS + s) * 32 + lane];
        v16bf b3 = bp[(3 * KS + s) * 32 + lane];
        acc0 = __builtin_amdgcn_wmma_f32_16x16x32_bf16(false, a, false, b0, (short)0, acc0, false, false);
        acc1 = __builtin_amdgcn_wmma_f32_16x16x32_bf16(false, a, false, b1, (short)0, acc1, false, false);
        acc2 = __builtin_amdgcn_wmma_f32_16x16x32_bf16(false, a, false, b2, (short)0, acc2, false, false);
        acc3 = __builtin_amdgcn_wmma_f32_16x16x32_bf16(false, a, false, b3, (short)0, acc3, false, false);
    }
    // C/D layout: VGPR v -> M = v + 8*(lane>=16); N = lane&15 (+16 per tile)
    const int rbase = mtile * 16 + ((lane >> 4) << 3);
    const int c     = lane & 15;
#pragma unroll
    for (int v = 0; v < 8; ++v) {
        float* hr = H + (size_t)(rbase + v) * HID;
        hr[c]      = acc0[v];
        hr[16 + c] = acc1[v];
        hr[32 + c] = acc2[v];
        hr[48 + c] = acc3[v];
    }
}

// ---------- per-node attention scores: As = h.a_src, Ad = h.a_dst ----------
__global__ void alpha_k(const float* __restrict__ H,
                        const float* __restrict__ asrc, const float* __restrict__ adst,
                        float* __restrict__ As, float* __restrict__ Ad, int N) {
    int n = blockIdx.x * blockDim.x + threadIdx.x;
    if (n >= N) return;
    const float4* h = (const float4*)(H + (size_t)n * HID);
    float s = 0.f, d = 0.f;
#pragma unroll
    for (int i = 0; i < 16; ++i) {
        float4 v = h[i];
        s += v.x * asrc[4*i] + v.y * asrc[4*i+1] + v.z * asrc[4*i+2] + v.w * asrc[4*i+3];
        d += v.x * adst[4*i] + v.y * adst[4*i+1] + v.z * adst[4*i+2] + v.w * adst[4*i+3];
    }
    As[n] = s; Ad[n] = d;
}

// ---------- init: m = enc(-inf), den = 0, agg = 0 ----------
__global__ void init_k(unsigned* __restrict__ menc, float* __restrict__ den,
                       float* __restrict__ agg, int N) {
    int i = blockIdx.x * blockDim.x + threadIdx.x;
    if (i < N) { menc[i] = 0x007FFFFFu; den[i] = 0.0f; }   // enc(-inf)
    if (i < N * HID) agg[i] = 0.0f;
}

// ---------- segment max over incoming edges ----------
__global__ void edge_max_k(const int* __restrict__ ei, int E, int N,
                           const float* __restrict__ As, const float* __restrict__ Ad,
                           unsigned* __restrict__ menc) {
    int e = blockIdx.x * blockDim.x + threadIdx.x;
    if (e >= E + N) return;
    int s, d; edge_sd(ei, E, e, s, d);
    atomicMax(&menc[d], fenc(edge_e(As, Ad, s, d)));
}

// ---------- segment sum of exp(e - max) ----------
__global__ void edge_sum_k(const int* __restrict__ ei, int E, int N,
                           const float* __restrict__ As, const float* __restrict__ Ad,
                           const unsigned* __restrict__ menc, float* __restrict__ den) {
    int e = blockIdx.x * blockDim.x + threadIdx.x;
    if (e >= E + N) return;
    int s, d; edge_sd(ei, E, e, s, d);
    float w = __expf(edge_e(As, Ad, s, d) - fdec(menc[d]));
    atomicAdd(&den[d], w);
}

// ---------- weighted aggregation: agg[dst] += alpha * h[src], 16 threads/edge ----------
__global__ void edge_aggr_k(const int* __restrict__ ei, int E, int N,
                            const float* __restrict__ As, const float* __restrict__ Ad,
                            const unsigned* __restrict__ menc, const float* __restrict__ den,
                            const float* __restrict__ H, float* __restrict__ agg) {
    long long id = (long long)blockIdx.x * blockDim.x + threadIdx.x;
    int e = (int)(id >> 4);
    int t = (int)(id & 15);
    if (e >= E + N) return;
    int s, d; edge_sd(ei, E, e, s, d);
    float w = __expf(edge_e(As, Ad, s, d) - fdec(menc[d]));
    float alpha = w / den[d];
    float4 hv = *(const float4*)(H + (size_t)s * HID + t * 4);
    float* o = agg + (size_t)d * HID + t * 4;
    atomicAdd(o + 0, alpha * hv.x);
    atomicAdd(o + 1, alpha * hv.y);
    atomicAdd(o + 2, alpha * hv.z);
    atomicAdd(o + 3, alpha * hv.w);
}

// ---------- out = relu(agg + b) ----------
__global__ void bias_relu_k(float* __restrict__ agg, const float* __restrict__ b, int N) {
    int i = blockIdx.x * blockDim.x + threadIdx.x;
    if (i >= N * HID) return;
    float v = agg[i] + b[i & (HID - 1)];
    agg[i] = v > 0.0f ? v : 0.0f;
}

// ---------- heads: factors [N,3] then skills [N,7], concatenated flat ----------
__global__ void heads_k(const float* __restrict__ H,
                        const float* __restrict__ Wf, const float* __restrict__ bf_,
                        const float* __restrict__ Ws, const float* __restrict__ bs_,
                        float* __restrict__ out, int N) {
    int n = blockIdx.x * blockDim.x + threadIdx.x;
    if (n >= N) return;
    const float* h = H + (size_t)n * HID;
    float f[3], sk[7];
#pragma unroll
    for (int j = 0; j < 3; ++j) f[j] = bf_[j];
#pragma unroll
    for (int j = 0; j < 7; ++j) sk[j] = bs_[j];
    for (int i = 0; i < HID; ++i) {
        float v = h[i];
#pragma unroll
        for (int j = 0; j < 3; ++j) f[j] += v * Wf[i * 3 + j];
#pragma unroll
        for (int j = 0; j < 7; ++j) sk[j] += v * Ws[i * 7 + j];
    }
#pragma unroll
    for (int j = 0; j < 3; ++j) out[(size_t)n * 3 + j] = f[j];
    float* so = out + (size_t)N * 3;
#pragma unroll
    for (int j = 0; j < 7; ++j) so[(size_t)n * 7 + j] = sk[j];
}

// ============================== launch ==============================
extern "C" void kernel_launch(void* const* d_in, const int* in_sizes, int n_in,
                              void* d_out, int out_size, void* d_ws, size_t ws_size,
                              hipStream_t stream) {
    const float* x   = (const float*)d_in[0];
    const int*   ei  = (const int*)d_in[1];
    const float* W1  = (const float*)d_in[2];
    const float* as1 = (const float*)d_in[3];
    const float* ad1 = (const float*)d_in[4];
    const float* b1  = (const float*)d_in[5];
    const float* W2  = (const float*)d_in[6];
    const float* as2 = (const float*)d_in[7];
    const float* ad2 = (const float*)d_in[8];
    const float* b2  = (const float*)d_in[9];
    const float* Wf  = (const float*)d_in[10];
    const float* bf_ = (const float*)d_in[11];
    const float* Ws  = (const float*)d_in[12];
    const float* bs_ = (const float*)d_in[13];
    float* out = (float*)d_out;

    const int IN_CH = 256;
    const int N  = in_sizes[0] / IN_CH;   // 100000
    const int E  = in_sizes[1] / 2;       // 1600000
    const int ET = E + N;                 // edges + self-loops

    // workspace layout (f32 units unless noted)
    float*    H1   = (float*)d_ws;                 // N*64  GEMM output / layer-2 features
    float*    HA   = H1 + (size_t)N * HID;         // N*64  aggregation target / layer-2 input
    float*    As   = HA + (size_t)N * HID;         // N
    float*    Ad   = As + N;                       // N
    unsigned* menc = (unsigned*)(Ad + N);          // N
    float*    den  = (float*)(menc + N);           // N
    // 32B-align packed bf16 weights
    size_t woff = (((size_t)(den + N) - (size_t)d_ws) + 31) & ~(size_t)31;
    unsigned short* Wp1 = (unsigned short*)((char*)d_ws + woff);   // 4*8*32*16 bf16 = 32KB
    unsigned short* Wp2 = Wp1 + 4 * 8 * 32 * 16;                   // 4*2*32*16 bf16 = 8KB

    const int B = 256;
    const int gN     = (N + B - 1) / B;
    const int gNH    = (N * HID + B - 1) / B;
    const int gE     = (ET + B - 1) / B;
    const int gE16   = (int)(((long long)ET * 16 + B - 1) / B);
    const int waves  = (N + 15) / 16;
    const int gGemm  = (waves * 32 + B - 1) / B;

    // pack both weight matrices to bf16 fragments
    pack_w_bf16<<<(4 * 8 * 32 + B - 1) / B, B, 0, stream>>>(W1, (__bf16*)Wp1, IN_CH);
    pack_w_bf16<<<(4 * 2 * 32 + B - 1) / B, B, 0, stream>>>(W2, (__bf16*)Wp2, HID);

    // ---- layer 1 ----
    gat_gemm_bf16<<<gGemm, B, 0, stream>>>(x, (const __bf16*)Wp1, H1, N, IN_CH);
    alpha_k<<<gN, B, 0, stream>>>(H1, as1, ad1, As, Ad, N);
    init_k<<<gNH, B, 0, stream>>>(menc, den, HA, N);
    edge_max_k<<<gE, B, 0, stream>>>(ei, E, N, As, Ad, menc);
    edge_sum_k<<<gE, B, 0, stream>>>(ei, E, N, As, Ad, menc, den);
    edge_aggr_k<<<gE16, B, 0, stream>>>(ei, E, N, As, Ad, menc, den, H1, HA);
    bias_relu_k<<<gNH, B, 0, stream>>>(HA, b1, N);

    // ---- layer 2 ----
    gat_gemm_bf16<<<gGemm, B, 0, stream>>>(HA, (const __bf16*)Wp2, H1, N, HID);
    alpha_k<<<gN, B, 0, stream>>>(H1, as2, ad2, As, Ad, N);
    init_k<<<gNH, B, 0, stream>>>(menc, den, HA, N);   // HA reused as agg target
    edge_max_k<<<gE, B, 0, stream>>>(ei, E, N, As, Ad, menc);
    edge_sum_k<<<gE, B, 0, stream>>>(ei, E, N, As, Ad, menc, den);
    edge_aggr_k<<<gE16, B, 0, stream>>>(ei, E, N, As, Ad, menc, den, H1, HA);
    bias_relu_k<<<gNH, B, 0, stream>>>(HA, b2, N);

    // ---- heads ----
    heads_k<<<gN, B, 0, stream>>>(HA, Wf, bf_, Ws, bs_, out, N);
}
